// Attention_layer_78847009620188
// MI455X (gfx1250) — matrix operand order repaired and checked by reading, not compiled, not run on hardware
//
#include <hip/hip_runtime.h>

// MI455X / gfx1250, wave32. bf16 WMMA (16x16x32, f32 accum) everywhere.

typedef __attribute__((ext_vector_type(16))) __bf16 v16bf;
typedef __attribute__((ext_vector_type(2)))  __bf16 bf16x2;
typedef __attribute__((ext_vector_type(8)))  float  v8f;

constexpr int kC    = 128;     // in channels == Cs
constexpr int kD    = 32;      // depth (attention sequence axis)
constexpr int kHW   = 4096;    // 64*64
constexpr int kNPix = 8192;    // B * HW
constexpr int kCD   = kC * kD; // 4096 elements per pixel panel

static __device__ __forceinline__ v8f wmma_bf16(v16bf a, v16bf b, v8f c) {
  return __builtin_amdgcn_wmma_f32_16x16x32_bf16(false, a, false, b, (short)0, c,
                                                 false, false);
}

// pack two f32 -> packed bf16x2 dword (targets v_cvt_pk_bf16_f32)
static __device__ __forceinline__ unsigned int pk_bf16(float a, float b) {
  bf16x2 t;
  t.x = (__bf16)a;
  t.y = (__bf16)b;
  return __builtin_bit_cast(unsigned int, t);
}

// A-matrix 16x32 bf16 fragment from a row-major bf16 panel (LDS or global).
// lane<16 : row = base row + (lane&15), elements = K {k..k+7, k+16..k+23}
// lane>=16: same row,                elements = K {k+8..k+15, k+24..k+31}
static __device__ __forceinline__ v16bf
load_a_frag(const char* base, int rowBytes, int row, int kByteOff, int lane) {
  const char* p = base + row * rowBytes + kByteOff + ((lane & 16) ? 16 : 0);
  union { v16bf v; float4 q[2]; } u;
  u.q[0] = *(const float4*)(p);
  u.q[1] = *(const float4*)(p + 32);
  return u.v;
}

// pack 8 f32 accumulator rows -> 8 bf16 (16B contiguous store)
static __device__ __forceinline__ void store_row8_bf16(char* dst, v8f a) {
  uint4 q;
  q.x = pk_bf16(a[0], a[1]);
  q.y = pk_bf16(a[2], a[3]);
  q.z = pk_bf16(a[4], a[5]);
  q.w = pk_bf16(a[6], a[7]);
  *(uint4*)dst = q;
}

// ---------------------------------------------------------------------------
// Kernel 0: convert the four 128x128 f32 weight matrices to bf16 in ws.
// Layout: wbf[mat][o][c], row = 256B. 64 blocks x 256 threads.
// ---------------------------------------------------------------------------
__global__ __launch_bounds__(256) void
k_prep_weights(const float* __restrict__ Wk, const float* __restrict__ Wq,
               const float* __restrict__ Wv, const float* __restrict__ Wo,
               __bf16* __restrict__ wbf) {
  const int i = blockIdx.x * 256 + threadIdx.x;   // 0..16383
  const float* wg[4] = {Wk, Wq, Wv, Wo};
#pragma unroll
  for (int m = 0; m < 4; ++m) wbf[m * 16384 + i] = (__bf16)wg[m][i];
}

// ---------------------------------------------------------------------------
// Kernel 1: x[b,c,d,hw] f32 (channel-major)  ->  xt[pix][d][c] bf16, pix=b*HW+hw
// Coalesced reads, LDS-staged contiguous writes. 16 pixels / block.
// ---------------------------------------------------------------------------
__global__ __launch_bounds__(256) void k_transpose_in(const float* __restrict__ x,
                                                      __bf16* __restrict__ xt) {
  extern __shared__ char smem[];            // 16 * 4096 * 2B = 128 KB
  __bf16* tile = (__bf16*)smem;             // [hw16][d*128+c]
  const int t    = threadIdx.x;
  const int pix0 = blockIdx.x * 16;
  const int b    = pix0 >> 12;
  const int hw0  = pix0 & (kHW - 1);
  const float* xb = x + (size_t)b * kC * kD * kHW;
  const int hw = t & 15;
#pragma unroll 4
  for (int it = 0; it < 256; ++it) {
    int cd = it * 16 + (t >> 4);            // c*32+d, 0..4095
    int c = cd >> 5, d = cd & 31;
    float v = xb[((size_t)c * kD + d) * kHW + hw0 + hw];
    tile[hw * kCD + d * kC + c] = (__bf16)v;
  }
  __syncthreads();
  const uint4* src = (const uint4*)smem;    // 8192 uint4
  uint4* dst = (uint4*)(xt + (size_t)pix0 * kCD);
#pragma unroll
  for (int i = 0; i < 32; ++i) dst[i * 256 + t] = src[i * 256 + t];
}

// ---------------------------------------------------------------------------
// Kernel 2: fused per-pixel attention. 4 waves / block, 1 pixel / wave.
// Weights read as A-fragments straight from global bf16 (L0/L2-served, huge
// reuse). LDS = 4 x 26KB wave-private panels only -> 3 WGs (12 waves) / WGP.
// No block barrier needed: all LDS traffic is wave-private.
// ---------------------------------------------------------------------------
__global__ __launch_bounds__(128) void
k_fused_attn(const __bf16* __restrict__ wbf,
             const float* __restrict__ bk, const float* __restrict__ bq,
             const float* __restrict__ bv, const float* __restrict__ bo,
             const __bf16* __restrict__ xt, float* __restrict__ ot) {
  extern __shared__ char smem[];
  const int t = threadIdx.x, lane = t & 31, wave = t >> 5;

  const char* wb  = (const char*)wbf;      // [mat][128][128] bf16, row = 256B
  const char* sWo = wb + 98304;

  // ---- per-wave private LDS panels (26624 B each) ----
  char* sc = smem + wave * 26624;
  char* Kp = sc;            // [32 d][128 c] bf16 (row 256B)  : K^T for scores-A
  char* Qp = sc + 8192;     // [32 d][128 c] bf16 (row 256B)  : scores-B; reused as Ap
  char* Vp = sc + 16384;    // [128 c][32 d] bf16             : AV-A
  char* Pp = sc + 24576;    // [32 j][32 i] bf16 (row 64B)    : softmax probs, AV-B

  const int pix  = blockIdx.x * 4 + wave;
  const int col  = lane & 15;
  const int rofs = (lane & 16) ? 8 : 0;    // C-fragment row split across lane halves

  // ---- resident X B-fragments: [32c x 16d], lane = d column ----
  const char* xp = (const char*)(xt + (size_t)pix * kCD);
  v16bf xf[4][2];
#pragma unroll
  for (int kc = 0; kc < 4; ++kc)
#pragma unroll
    for (int nh = 0; nh < 2; ++nh)
      xf[kc][nh] = load_a_frag(xp, 256, nh * 16 + col, kc * 64, lane);

  // ---- K, Q, V = W * X + b (192 WMMAs / wave) ----
#pragma unroll 2
  for (int mt = 0; mt < 8; ++mt) {
    const int ob = mt * 16 + rofs;         // first of this lane-half's 8 output rows
#pragma unroll
    for (int m = 0; m < 3; ++m) {
      const char*  w  = wb + m * 32768;
      const float* bp = ((m == 0) ? bk : (m == 1) ? bq : bv) + ob;
      float4 b0 = *(const float4*)(bp), b1 = *(const float4*)(bp + 4);
      v8f acc[2];
#pragma unroll
      for (int nh = 0; nh < 2; ++nh) {
        acc[nh][0] = b0.x; acc[nh][1] = b0.y; acc[nh][2] = b0.z; acc[nh][3] = b0.w;
        acc[nh][4] = b1.x; acc[nh][5] = b1.y; acc[nh][6] = b1.z; acc[nh][7] = b1.w;
      }
#pragma unroll
      for (int kc = 0; kc < 4; ++kc) {
        v16bf a = load_a_frag(w, 256, mt * 16 + col, kc * 64, lane);
        acc[0] = wmma_bf16(a, xf[kc][0], acc[0]);
        acc[1] = wmma_bf16(a, xf[kc][1], acc[1]);
      }
      if (m == 0) {                         // K -> Kp[d][c] (16B contiguous)
#pragma unroll
        for (int nh = 0; nh < 2; ++nh)
          store_row8_bf16(Kp + (nh * 16 + col) * 256 + ob * 2, acc[nh]);
      } else {                              // Q -> [d][c], V -> [c][d] (per-element scatter)
        __bf16* dst = (__bf16*)((m == 1) ? Qp : Vp);
#pragma unroll
        for (int nh = 0; nh < 2; ++nh) {
          int d = nh * 16 + col;
#pragma unroll
          for (int r = 0; r < 8; ++r) dst[(m == 1) ? d * 128 + ob + r : (ob + r) * 32 + d] = (__bf16)acc[nh][r];
        }
      }
    }
  }

  // ---- scores = (K^T Q) * scale ; softmax over key axis i ; P -> Pp ----
  const float scale = 0.08838834764831845f;   // 1/sqrt(128)
#pragma unroll
  for (int jt = 0; jt < 2; ++jt) {
    v8f s0 = {0, 0, 0, 0, 0, 0, 0, 0};        // it = 0 (i rows rofs..rofs+7)
    v8f s1 = {0, 0, 0, 0, 0, 0, 0, 0};        // it = 1 (i rows 16+rofs..)
#pragma unroll
    for (int kc = 0; kc < 4; ++kc) {
      v16bf bqf = load_a_frag(Qp, 256, jt * 16 + col, kc * 64, lane);
      v16bf a0  = load_a_frag(Kp, 256, col,      kc * 64, lane);
      v16bf a1  = load_a_frag(Kp, 256, 16 + col, kc * 64, lane);
      s0 = wmma_bf16(a0, bqf, s0);
      s1 = wmma_bf16(a1, bqf, s1);
    }
    float e0[8], e1[8], mloc = -3.4e38f;
#pragma unroll
    for (int r = 0; r < 8; ++r) {
      e0[r] = s0[r] * scale; mloc = fmaxf(mloc, e0[r]);
      e1[r] = s1[r] * scale; mloc = fmaxf(mloc, e1[r]);
    }
    mloc = fmaxf(mloc, __shfl_xor(mloc, 16, 32));   // other half of i rows
    float ssum = 0.f;
#pragma unroll
    for (int r = 0; r < 8; ++r) {
      e0[r] = __expf(e0[r] - mloc); ssum += e0[r];
      e1[r] = __expf(e1[r] - mloc); ssum += e1[r];
    }
    ssum += __shfl_xor(ssum, 16, 32);
    float inv = 1.0f / ssum;
    __bf16* pp = (__bf16*)Pp;
    int j = jt * 16 + col;
#pragma unroll
    for (int r = 0; r < 8; ++r) {
      pp[j * 32 + rofs + r]      = (__bf16)(e0[r] * inv);
      pp[j * 32 + 16 + rofs + r] = (__bf16)(e1[r] * inv);
    }
  }

  // ---- a = V * P  -> Ap (reuse Qp)  (16 WMMAs) ----
  __bf16* Ap = (__bf16*)Qp;                  // Q panel dead after scores
#pragma unroll
  for (int mt = 0; mt < 8; ++mt) {
    v16bf a = load_a_frag(Vp, 64, mt * 16 + col, 0, lane);   // [16c x 32i]
#pragma unroll
    for (int jt = 0; jt < 2; ++jt) {
      v16bf b = load_a_frag(Pp, 64, jt * 16 + col, 0, lane);   // [32i x 16j]
      v8f acc = {0, 0, 0, 0, 0, 0, 0, 0};
      acc = wmma_bf16(a, b, acc);
      int d = jt * 16 + col;
#pragma unroll
      for (int r = 0; r < 8; ++r) Ap[d * 128 + mt * 16 + rofs + r] = (__bf16)acc[r];
    }
  }

  // ---- delta = Wo * a + bo  -> ot[pix][d][o] f32 (64 WMMAs) ----
  float* op = ot + (size_t)pix * kCD;
#pragma unroll 2
  for (int mt = 0; mt < 8; ++mt) {
    const int ob = mt * 16 + rofs;
    const float* bp = bo + ob;
    float4 b0 = *(const float4*)(bp), b1 = *(const float4*)(bp + 4);
    v8f acc[2];
#pragma unroll
    for (int nh = 0; nh < 2; ++nh) {
      acc[nh][0] = b0.x; acc[nh][1] = b0.y; acc[nh][2] = b0.z; acc[nh][3] = b0.w;
      acc[nh][4] = b1.x; acc[nh][5] = b1.y; acc[nh][6] = b1.z; acc[nh][7] = b1.w;
    }
#pragma unroll
    for (int kc = 0; kc < 4; ++kc) {
      v16bf a = load_a_frag(sWo, 256, mt * 16 + col, kc * 64, lane);
#pragma unroll
      for (int nh = 0; nh < 2; ++nh) {
        v16bf b = load_a_frag((const char*)Ap, 256, nh * 16 + col, kc * 64, lane);
        acc[nh] = wmma_bf16(a, b, acc[nh]);
      }
    }
#pragma unroll
    for (int nh = 0; nh < 2; ++nh) {
      int d = nh * 16 + col;
      float* dst = op + d * kC + ob;         // 32B-aligned contiguous o-run
      float4 lo; lo.x = acc[nh][0]; lo.y = acc[nh][1]; lo.z = acc[nh][2]; lo.w = acc[nh][3];
      float4 hi; hi.x = acc[nh][4]; hi.y = acc[nh][5]; hi.z = acc[nh][6]; hi.w = acc[nh][7];
      *(float4*)dst       = lo;
      *(float4*)(dst + 4) = hi;
    }
  }
}

// ---------------------------------------------------------------------------
// Kernel 3: out[b,o,d,hw] = x + ot[pix][d][o]  (LDS-staged un-transpose)
// 8 pixels / block.
// ---------------------------------------------------------------------------
__global__ __launch_bounds__(256) void
k_untranspose_add(const float* __restrict__ x, const float* __restrict__ ot,
                  float* __restrict__ out) {
  extern __shared__ char smem[];            // 8 * 4096 * 4B = 128 KB
  float* tile = (float*)smem;               // [hw8][d*128+o]
  const int t    = threadIdx.x;
  const int pix0 = blockIdx.x * 8;
  const int b    = pix0 >> 12;
  const int hw0  = pix0 & (kHW - 1);
  const uint4* src = (const uint4*)(ot + (size_t)pix0 * kCD);
  uint4* d4 = (uint4*)smem;
#pragma unroll
  for (int i = 0; i < 32; ++i) d4[i * 256 + t] = src[i * 256 + t];   // 8192 uint4
  __syncthreads();
  const float* xb = x   + (size_t)b * kC * kD * kHW;
  float*       ob = out + (size_t)b * kC * kD * kHW;
  const int hw = t & 7, sub = t >> 3;
#pragma unroll 4
  for (int it = 0; it < 128; ++it) {
    int od = it * 32 + sub;                 // o*32+d
    int o = od >> 5, d = od & 31;
    size_t g = ((size_t)o * kD + d) * kHW + hw0 + hw;
    ob[g] = xb[g] + tile[hw * kCD + d * kC + o];
  }
}

// ---------------------------------------------------------------------------
extern "C" void kernel_launch(void* const* d_in, const int* in_sizes, int n_in,
                              void* d_out, int out_size, void* d_ws, size_t ws_size,
                              hipStream_t stream) {
  (void)in_sizes; (void)n_in; (void)out_size; (void)ws_size;
  const float* x  = (const float*)d_in[0];
  const float* Wk = (const float*)d_in[1];
  const float* bk = (const float*)d_in[2];
  const float* Wq = (const float*)d_in[3];
  const float* bq = (const float*)d_in[4];
  const float* Wv = (const float*)d_in[5];
  const float* bv = (const float*)d_in[6];
  const float* Wo = (const float*)d_in[7];
  const float* bo = (const float*)d_in[8];

  __bf16* wbf = (__bf16*)d_ws;                                        // 128 KB
  __bf16* xt  = (__bf16*)((char*)d_ws + 131072);                      // 64 MB
  float*  ot  = (float*)((char*)d_ws + 131072 +
                         (size_t)kNPix * kCD * 2);                    // 128 MB
  float*  out = (float*)d_out;

  k_prep_weights<<<64, 256, 0, stream>>>(Wk, Wq, Wv, Wo, wbf);
  k_transpose_in<<<kNPix / 16, 256, 131072, stream>>>(x, xt);
  k_fused_attn<<<kNPix / 4, 128, 106496, stream>>>(wbf, bk, bq, bv, bo, xt, ot);
  k_untranspose_add<<<kNPix / 8, 256, 131072, stream>>>(x, ot, out);
}
